// MultiHeadLatentAttention_72430328480102
// MI455X (gfx1250) — matrix operand order repaired and checked by reading, not compiled, not run on hardware
//
#include <hip/hip_runtime.h>
#include <hip/hip_bf16.h>

typedef __bf16 bf16_t;
typedef __attribute__((ext_vector_type(16))) __bf16 v16bf;
typedef __attribute__((ext_vector_type(8)))  __bf16 v8bf;
typedef __attribute__((ext_vector_type(8)))  float  v8f;

static constexpr int Bb  = 2;
static constexpr int Tt  = 2048;
static constexpr int E   = 1024;
static constexpr int H   = 16;
static constexpr int DH  = 64;
static constexpr int DKV = 256;
static constexpr int DR  = 32;
static constexpr int DQK = DH + DR;           // 96
static constexpr float QSCALE = 0.10206207261596575f;  // 1/sqrt(96)
static constexpr float LN1E4_OVER_16 = 0.57564627324851148f; // ln(10000)/16

union BF16x16 { v16bf v; v8bf h[2]; };

__device__ __forceinline__ v8bf load8(const bf16_t* p) { return *(const v8bf*)p; }

// ---------------------------------------------------------------------------
// gfx1250 async global->LDS copy (ASYNCcnt engine).
// Probed signature: (v4i AS(1)*, v4i AS(3)*, int offset, int cpol).
// ---------------------------------------------------------------------------
#if __has_builtin(__builtin_amdgcn_global_load_async_to_lds_b128)
#define HAVE_ASYNC_LDS 1
#endif

typedef int v4i __attribute__((vector_size(16)));
typedef __attribute__((address_space(1))) v4i* g_v4i_p;
typedef __attribute__((address_space(3))) v4i* l_v4i_p;

__device__ __forceinline__ void async_cp16(const bf16_t* g, bf16_t* l) {
#ifdef HAVE_ASYNC_LDS
    // Generic LDS address low 32 bits == LDS byte offset (ISA 10.2); generic
    // global address bits == AS(1) address.
    __builtin_amdgcn_global_load_async_to_lds_b128(
        (g_v4i_p)(uintptr_t)g,
        (l_v4i_p)(uint32_t)(uintptr_t)l, 0, 0);
#else
    *(v8bf*)l = *(const v8bf*)g;
#endif
}

__device__ __forceinline__ void wait_async() {
#if __has_builtin(__builtin_amdgcn_s_wait_asynccnt)
    __builtin_amdgcn_s_wait_asynccnt(0);
#else
    asm volatile("s_wait_asynccnt 0x0" ::: "memory");
#endif
}

// ---------------------------------------------------------------------------
// Elementwise convert / transpose-convert
// ---------------------------------------------------------------------------
__global__ void cvt_f32_bf16(const float* __restrict__ s, bf16_t* __restrict__ d, int n) {
    int i = blockIdx.x * blockDim.x + threadIdx.x;
    if (i < n) d[i] = (bf16_t)s[i];
}

__global__ void transpose_cvt(const float* __restrict__ s, bf16_t* __restrict__ d, int K, int N) {
    int i = blockIdx.x * blockDim.x + threadIdx.x;
    if (i < K * N) {
        int k = i / N, n = i % N;
        d[(size_t)n * K + k] = (bf16_t)s[i];
    }
}

// ---------------------------------------------------------------------------
// bf16 WMMA GEMM: C(MxN) = A(MxK) * B(KxN), B given transposed [n][k].
// One wave computes a 16 x (16*NT) strip: A fragment reused NT times.
// MODE: 0=bf16 row-major, 1=f32 row-major, 2=transposed store into
//       vT[b][h][dh][t] (bf16).
// ---------------------------------------------------------------------------
template <int MODE, int NT>
__global__ __launch_bounds__(32)
void gemm16(const bf16_t* __restrict__ A, const bf16_t* __restrict__ BT,
            void* __restrict__ Cv, int M, int N, int K) {
    const int lane = threadIdx.x;
    const int half = lane >> 4;
    const int ln   = lane & 15;
    const int tilesN = N / (16 * NT);
    const int tm = blockIdx.x / tilesN;
    const int tn = blockIdx.x % tilesN;

    const bf16_t* arow = A + (size_t)(tm * 16 + ln) * K;
    const bf16_t* bcol[NT];
    #pragma unroll
    for (int j = 0; j < NT; j++)
        bcol[j] = BT + (size_t)((tn * NT + j) * 16 + ln) * K;

    v8f acc[NT];
    #pragma unroll
    for (int j = 0; j < NT; j++) acc[j] = (v8f){};

    for (int kb = 0; kb < K; kb += 32) {
        BF16x16 a;
        a.h[0] = load8(arow + kb + 8 * half);
        a.h[1] = load8(arow + kb + 16 + 8 * half);
        if (kb + 32 < K) __builtin_prefetch(arow + kb + 32, 0, 0);
        #pragma unroll
        for (int j = 0; j < NT; j++) {
            BF16x16 b;
            b.h[0] = load8(bcol[j] + kb + 8 * half);
            b.h[1] = load8(bcol[j] + kb + 16 + 8 * half);
            if (kb + 32 < K) __builtin_prefetch(bcol[j] + kb + 32, 0, 0);
            acc[j] = __builtin_amdgcn_wmma_f32_16x16x32_bf16(false, a.v, false, b.v,
                                                             (short)0, acc[j], false, false);
        }
    }

    #pragma unroll
    for (int j = 0; j < NT; j++) {
        const int n = (tn * NT + j) * 16 + ln;
        if constexpr (MODE == 0) {
            bf16_t* C = (bf16_t*)Cv;
            #pragma unroll
            for (int r = 0; r < 8; r++)
                C[(size_t)(tm * 16 + 8 * half + r) * N + n] = (bf16_t)acc[j][r];
        } else if constexpr (MODE == 1) {
            float* C = (float*)Cv;
            #pragma unroll
            for (int r = 0; r < 8; r++)
                C[(size_t)(tm * 16 + 8 * half + r) * N + n] = acc[j][r];
        } else {
            // vT[((b*H + n/64)*DH + n%64)*Tt + t], rows m = b*Tt + t
            bf16_t* C = (bf16_t*)Cv;
            const int m0 = tm * 16 + 8 * half;
            const int bi = m0 >> 11;          // / Tt
            const int t0 = m0 & (Tt - 1);
            v8bf pk;
            #pragma unroll
            for (int r = 0; r < 8; r++) pk[r] = (bf16_t)acc[j][r];
            *(v8bf*)(C + ((size_t)((bi * H + (n >> 6)) * DH + (n & 63))) * Tt + t0) = pk;
        }
    }
}

// ---------------------------------------------------------------------------
// RoPE + pack kernels.  q[b][h][t][96] = [q_c | rope(q_r)] * QSCALE  (bf16)
//                       k[b][h][t][96] = [k_c | rope(k_r) broadcast] (bf16)
// ---------------------------------------------------------------------------
__global__ void pack_q(const bf16_t* __restrict__ qc, const float* __restrict__ qr,
                       bf16_t* __restrict__ q) {
    int idx = blockIdx.x * blockDim.x + threadIdx.x;   // = (b*H + h)*Tt + t
    if (idx >= Bb * H * Tt) return;
    const int t = idx % Tt;
    const int h = (idx / Tt) % H;
    const int b = idx / (Tt * H);
    const bf16_t* src = qc + (size_t)(b * Tt + t) * (H * DH) + h * DH;
    const float*  r   = qr + (size_t)(b * Tt + t) * (H * DR) + h * DR;
    bf16_t* dst = q + (size_t)idx * DQK;
    for (int j = 0; j < DH; j++) dst[j] = (bf16_t)((float)src[j] * QSCALE);
    for (int i = 0; i < 16; i++) {
        float sn, cs;
        float inv = __expf(-LN1E4_OVER_16 * (float)i);
        __sincosf((float)t * inv, &sn, &cs);
        float x1 = r[i], x2 = r[i + 16];
        dst[DH + i]      = (bf16_t)((x1 * cs - x2 * sn) * QSCALE);
        dst[DH + 16 + i] = (bf16_t)((x2 * cs + x1 * sn) * QSCALE);
    }
}

__global__ void pack_k(const bf16_t* __restrict__ kc, const float* __restrict__ kr,
                       bf16_t* __restrict__ kk) {
    int idx = blockIdx.x * blockDim.x + threadIdx.x;
    if (idx >= Bb * H * Tt) return;
    const int t = idx % Tt;
    const int h = (idx / Tt) % H;
    const int b = idx / (Tt * H);
    const bf16_t* src = kc + (size_t)(b * Tt + t) * (H * DH) + h * DH;
    const float*  r   = kr + (size_t)(b * Tt + t) * DR;
    bf16_t* dst = kk + (size_t)idx * DQK;
    for (int j = 0; j < DH; j++) dst[j] = src[j];
    for (int i = 0; i < 16; i++) {
        float sn, cs;
        float inv = __expf(-LN1E4_OVER_16 * (float)i);
        __sincosf((float)t * inv, &sn, &cs);
        float x1 = r[i], x2 = r[i + 16];
        dst[DH + i]      = (bf16_t)(x1 * cs - x2 * sn);
        dst[DH + 16 + i] = (bf16_t)(x2 * cs + x1 * sn);
    }
}

// ---------------------------------------------------------------------------
// Flash attention with causal mask + LOBO denominator.
// One wave handles 16 queries of one (b,h); 32 keys per step. K/V blocks are
// double-buffered in LDS via the gfx1250 async global->LDS engine, prefetching
// block kb+32 while WMMAs run on block kb.
// ---------------------------------------------------------------------------
__global__ __launch_bounds__(32)
void mla_attn(const bf16_t* __restrict__ q, const bf16_t* __restrict__ k,
              const bf16_t* __restrict__ vT, const float* __restrict__ lobo,
              bf16_t* __restrict__ y) {
    __shared__ alignas(16) bf16_t klds[2][32 * DQK];  // 2 x 6144 B
    __shared__ alignas(16) bf16_t vlds[2][DH * 32];   // 2 x 4096 B
    __shared__ alignas(16) bf16_t pst[2 * 16 * 16];   // P staging [subtile][row][col]

    const int lane = threadIdx.x;
    const int half = lane >> 4;
    const int ln   = lane & 15;
    const int qt = blockIdx.x % (Tt / 16);
    const int h  = (blockIdx.x / (Tt / 16)) % H;
    const int b  = blockIdx.x / ((Tt / 16) * H);
    const int q0 = qt * 16;

    const bf16_t* qbase = q  + ((size_t)(b * H + h) * Tt + q0) * DQK;
    const bf16_t* kbase = k  + ((size_t)(b * H + h) * Tt) * DQK;
    const bf16_t* vbase = vT + ((size_t)(b * H + h) * DH) * Tt;

    // Stage one 32-key block of K (32x96, contiguous) and V^T (64 rows x 64B).
    auto stage_block = [&](int kb, int buf) {
        const bf16_t* kg = kbase + (size_t)kb * DQK;
        #pragma unroll
        for (int i = 0; i < 12; i++) {            // 384 x 16B chunks / 32 lanes
            const int c = i * 32 + lane;
            async_cp16(kg + c * 8, &klds[buf][c * 8]);
        }
        #pragma unroll
        for (int i = 0; i < 8; i++) {             // 64 rows x 4 chunks
            const int c   = i * 32 + lane;
            const int row = c >> 2;
            const int col = (c & 3) * 8;
            async_cp16(vbase + (size_t)row * Tt + kb + col, &vlds[buf][row * 32 + col]);
        }
    };

    // Q fragments (A operand), 3 feature blocks of 32
    BF16x16 qa[3];
    const bf16_t* qrow = qbase + (size_t)ln * DQK;
    #pragma unroll
    for (int f = 0; f < 3; f++) {
        qa[f].h[0] = load8(qrow + f * 32 + 8 * half);
        qa[f].h[1] = load8(qrow + f * 32 + 16 + 8 * half);
    }

    v8f o0 = {}, o1 = {}, o2 = {}, o3 = {};
    float m[8], l[8];
    #pragma unroll
    for (int r = 0; r < 8; r++) { m[r] = -INFINITY; l[r] = 0.f; }

    const int kend = q0 + 16;
    int buf = 0;
    stage_block(0, 0);

    for (int kb = 0; kb < kend; kb += 32) {
        wait_async();
        __syncthreads();

        // Read this block's fragments out of LDS.
        BF16x16 kf[2][3], vf[4];
        #pragma unroll
        for (int st = 0; st < 2; st++)
            #pragma unroll
            for (int f = 0; f < 3; f++) {
                const bf16_t* kr = &klds[buf][(st * 16 + ln) * DQK + f * 32 + 8 * half];
                kf[st][f].h[0] = load8(kr);
                kf[st][f].h[1] = load8(kr + 16);
            }
        #pragma unroll
        for (int nt = 0; nt < 4; nt++) {
            const bf16_t* vr = &vlds[buf][(nt * 16 + ln) * 32 + 8 * half];
            vf[nt].h[0] = load8(vr);
            vf[nt].h[1] = load8(vr + 16);
        }

        // Prefetch next block into the other buffer while we compute.
        if (kb + 32 < kend) stage_block(kb + 32, buf ^ 1);

        // --- S = Q * K^T for 2 subtiles of 16 keys ---
        v8f s[2];
        #pragma unroll
        for (int st = 0; st < 2; st++) {
            v8f sacc = {};
            #pragma unroll
            for (int f = 0; f < 3; f++)
                sacc = __builtin_amdgcn_wmma_f32_16x16x32_bf16(false, qa[f].v, false, kf[st][f].v,
                                                               (short)0, sacc, false, false);
            s[st] = sacc;
        }

        // --- causal mask (only near diagonal) ---
        if (kb + 32 > q0) {
            #pragma unroll
            for (int r = 0; r < 8; r++) {
                const int qg = q0 + 8 * half + r;
                if (kb + ln      > qg) s[0][r] = -1e30f;
                if (kb + 16 + ln > qg) s[1][r] = -1e30f;
            }
        }

        // --- online softmax (rows are lane-local: 8*half + r) ---
        float mloc[8];
        #pragma unroll
        for (int r = 0; r < 8; r++) mloc[r] = fmaxf(s[0][r], s[1][r]);
        #pragma unroll
        for (int off = 1; off < 16; off <<= 1)
            #pragma unroll
            for (int r = 0; r < 8; r++) mloc[r] = fmaxf(mloc[r], __shfl_xor(mloc[r], off, 32));

        float alpha[8], ls[8], p0[8], p1[8];
        #pragma unroll
        for (int r = 0; r < 8; r++) {
            float mn = fmaxf(m[r], mloc[r]);
            alpha[r] = __expf(m[r] - mn);
            p0[r] = __expf(s[0][r] - mn);
            p1[r] = __expf(s[1][r] - mn);
            m[r] = mn;
            ls[r] = p0[r] + p1[r];
        }
        #pragma unroll
        for (int off = 1; off < 16; off <<= 1)
            #pragma unroll
            for (int r = 0; r < 8; r++) ls[r] += __shfl_xor(ls[r], off, 32);
        #pragma unroll
        for (int r = 0; r < 8; r++) l[r] = l[r] * alpha[r] + ls[r];
        #pragma unroll
        for (int r = 0; r < 8; r++) {
            o0[r] *= alpha[r]; o1[r] *= alpha[r]; o2[r] *= alpha[r]; o3[r] *= alpha[r];
        }

        // --- stage P (C layout) -> LDS -> A layout ---
        __syncthreads();
        #pragma unroll
        for (int r = 0; r < 8; r++) {
            const int row = 8 * half + r;
            pst[row * 16 + ln]       = (bf16_t)p0[r];
            pst[256 + row * 16 + ln] = (bf16_t)p1[r];
        }
        __syncthreads();
        BF16x16 pa;
        pa.h[0] = *(const v8bf*)&pst[ln * 16 + 8 * half];
        pa.h[1] = *(const v8bf*)&pst[256 + ln * 16 + 8 * half];

        // --- O += P * V ---
        o0 = __builtin_amdgcn_wmma_f32_16x16x32_bf16(false, pa.v, false, vf[0].v, (short)0, o0, false, false);
        o1 = __builtin_amdgcn_wmma_f32_16x16x32_bf16(false, pa.v, false, vf[1].v, (short)0, o1, false, false);
        o2 = __builtin_amdgcn_wmma_f32_16x16x32_bf16(false, pa.v, false, vf[2].v, (short)0, o2, false, false);
        o3 = __builtin_amdgcn_wmma_f32_16x16x32_bf16(false, pa.v, false, vf[3].v, (short)0, o3, false, false);

        buf ^= 1;
    }

    // --- LOBO normalize and store y[b][t][h*64 + d] (bf16) ---
    const float C = __expf(lobo[h]);
    #pragma unroll
    for (int r = 0; r < 8; r++) {
        const float inv = 1.0f / (l[r] + C);
        const size_t base = (size_t)(b * Tt + q0 + 8 * half + r) * (H * DH) + h * DH + ln;
        y[base + 0]  = (bf16_t)(o0[r] * inv);
        y[base + 16] = (bf16_t)(o1[r] * inv);
        y[base + 32] = (bf16_t)(o2[r] * inv);
        y[base + 48] = (bf16_t)(o3[r] * inv);
    }
}

// ---------------------------------------------------------------------------
extern "C" void kernel_launch(void* const* d_in, const int* in_sizes, int n_in,
                              void* d_out, int out_size, void* d_ws, size_t ws_size,
                              hipStream_t stream) {
    (void)in_sizes; (void)n_in; (void)out_size; (void)ws_size;
    const float* x    = (const float*)d_in[0];
    const float* Wq   = (const float*)d_in[1];
    const float* Wqr  = (const float*)d_in[2];
    const float* Wkr  = (const float*)d_in[3];
    const float* Wkvd = (const float*)d_in[4];
    const float* Wku  = (const float*)d_in[5];
    const float* Wvu  = (const float*)d_in[6];
    const float* Wo   = (const float*)d_in[7];
    const float* lobo = (const float*)d_in[8];

    char* p = (char*)d_ws;
    auto alloc = [&](size_t bytes) -> void* {
        void* r = (void*)p;
        p += (bytes + 255) & ~(size_t)255;
        return r;
    };

    const size_t MR = (size_t)Bb * Tt;            // 4096 rows
    bf16_t* xb    = (bf16_t*)alloc(MR * E * 2);
    bf16_t* WqT   = (bf16_t*)alloc((size_t)E * (H * DH) * 2);
    bf16_t* WqrT  = (bf16_t*)alloc((size_t)E * (H * DR) * 2);
    bf16_t* WkrT  = (bf16_t*)alloc((size_t)E * DR * 2);
    bf16_t* WkvdT = (bf16_t*)alloc((size_t)E * DKV * 2);
    bf16_t* WkuT  = (bf16_t*)alloc((size_t)DKV * (H * DH) * 2);
    bf16_t* WvuT  = (bf16_t*)alloc((size_t)DKV * (H * DH) * 2);
    bf16_t* WoT   = (bf16_t*)alloc((size_t)E * E * 2);
    bf16_t* qcRaw = (bf16_t*)alloc(MR * (H * DH) * 2);
    float*  qrRaw = (float*) alloc(MR * (H * DR) * 4);
    float*  krRaw = (float*) alloc(MR * DR * 4);
    bf16_t* lat   = (bf16_t*)alloc(MR * DKV * 2);
    bf16_t* kcRaw = (bf16_t*)alloc(MR * (H * DH) * 2);
    bf16_t* vTb   = (bf16_t*)alloc((size_t)Bb * H * DH * Tt * 2);
    bf16_t* qbuf  = (bf16_t*)alloc((size_t)Bb * H * Tt * DQK * 2);
    bf16_t* kbuf  = (bf16_t*)alloc((size_t)Bb * H * Tt * DQK * 2);
    bf16_t* ybuf  = (bf16_t*)alloc(MR * (H * DH) * 2);

    const int TB = 256;
    auto g1 = [&](size_t n) { return (unsigned)((n + TB - 1) / TB); };

    // 1. Convert inputs to bf16 (weights transposed to [n][k]).
    cvt_f32_bf16<<<g1(MR * E), TB, 0, stream>>>(x, xb, (int)(MR * E));
    transpose_cvt<<<g1((size_t)E * 1024), TB, 0, stream>>>(Wq,   WqT,   E, 1024);
    transpose_cvt<<<g1((size_t)E * 512),  TB, 0, stream>>>(Wqr,  WqrT,  E, 512);
    transpose_cvt<<<g1((size_t)E * 32),   TB, 0, stream>>>(Wkr,  WkrT,  E, 32);
    transpose_cvt<<<g1((size_t)E * 256),  TB, 0, stream>>>(Wkvd, WkvdT, E, 256);
    transpose_cvt<<<g1((size_t)DKV * 1024), TB, 0, stream>>>(Wku, WkuT, DKV, 1024);
    transpose_cvt<<<g1((size_t)DKV * 1024), TB, 0, stream>>>(Wvu, WvuT, DKV, 1024);
    transpose_cvt<<<g1((size_t)E * E),    TB, 0, stream>>>(Wo,   WoT,   E, E);

    // 2. Projection GEMMs (wave per 16x64 strip; 16x32 for the DR=32 GEMM).
    gemm16<0, 4><<<(4096 / 16) * (1024 / 64), 32, 0, stream>>>(xb, WqT,   qcRaw, 4096, 1024, 1024);
    gemm16<1, 4><<<(4096 / 16) * (512 / 64),  32, 0, stream>>>(xb, WqrT,  qrRaw, 4096, 512, 1024);
    gemm16<1, 2><<<(4096 / 16) * (32 / 32),   32, 0, stream>>>(xb, WkrT,  krRaw, 4096, 32, 1024);
    gemm16<0, 4><<<(4096 / 16) * (256 / 64),  32, 0, stream>>>(xb, WkvdT, lat,   4096, 256, 1024);
    gemm16<0, 4><<<(4096 / 16) * (1024 / 64), 32, 0, stream>>>(lat, WkuT,  kcRaw, 4096, 1024, 256);
    gemm16<2, 4><<<(4096 / 16) * (1024 / 64), 32, 0, stream>>>(lat, WvuT,  vTb,   4096, 1024, 256);

    // 3. RoPE + pack into attention layouts.
    pack_q<<<g1((size_t)Bb * H * Tt), TB, 0, stream>>>(qcRaw, qrRaw, qbuf);
    pack_k<<<g1((size_t)Bb * H * Tt), TB, 0, stream>>>(kcRaw, krRaw, kbuf);

    // 4. Causal flash attention with LOBO denominator (async K/V staging).
    mla_attn<<<Bb * H * (Tt / 16), 32, 0, stream>>>(qbuf, kbuf, vTb, lobo, ybuf);

    // 5. Output projection -> f32 d_out.
    gemm16<1, 4><<<(4096 / 16) * (1024 / 64), 32, 0, stream>>>(ybuf, WoT, (float*)d_out,
                                                               4096, 1024, 1024);
}
